// ResBlockDownSparse_13030930776771
// MI455X (gfx1250) — matrix operand order repaired and checked by reading, not compiled, not run on hardware
//
#include <hip/hip_runtime.h>
#include <hip/hip_bf16.h>

typedef __attribute__((ext_vector_type(16))) __bf16         v16bf;
typedef __attribute__((ext_vector_type(8)))  float          v8f;
typedef __attribute__((ext_vector_type(16))) unsigned short us16;
typedef __attribute__((ext_vector_type(8)))  unsigned short us8;

#define EPS 1e-5f

// ---------- workspace layout (bytes, 256-aligned) ----------
// xm padded: [4][258][258][64] bf16, y1 padded: [4][130][130][128] bf16
#define OFF_XM    ((size_t)0)           // 34,080,768
#define OFF_RAW   ((size_t)34080768)    // f32 [4,128,128,128] 33,554,432 (reused)
#define OFF_Y1    ((size_t)67635200)    // 17,305,600
#define OFF_M1    ((size_t)84940800)    // u8 [4,128,128] 65,536
#define OFF_W1P   ((size_t)85006336)    // bf16 [18][8][32][16] 147,456
#define OFF_W2P   ((size_t)85153792)    // bf16 [36][8][32][16] 294,912
#define OFF_PSUM  ((size_t)85448704)    // f32 [1024][128] 524,288
#define OFF_PSQ   ((size_t)85972992)    // f32 [1024][128] 524,288
#define OFF_PCNT  ((size_t)86497280)    // f32 [1024] 4,096
#define OFF_SS    ((size_t)86501376)    // f32 [256] 1,024

__device__ __forceinline__ unsigned short f2bf_bits(float f) {
    unsigned u = __builtin_bit_cast(unsigned, f);
    unsigned r = u + 0x7FFFu + ((u >> 16) & 1u);   // RNE
    return (unsigned short)(r >> 16);
}

__device__ __forceinline__ v16bf mk_a(const unsigned short* p) {
    us8 s0 = *(const us8*)p;
    us8 s1 = *(const us8*)(p + 16);
    us16 abits;
    #pragma unroll
    for (int i = 0; i < 8; ++i) { abits[i] = s0[i]; abits[8 + i] = s1[i]; }
    return __builtin_bit_cast(v16bf, abits);
}

// ---------- prep kernels ----------
__global__ __launch_bounds__(256) void k_zero16(us8* __restrict__ p, int nvec) {
    int idx = blockIdx.x * 256 + threadIdx.x;
    if (idx < nvec) {
        us8 z;
        #pragma unroll
        for (int i = 0; i < 8; ++i) z[i] = 0;
        p[idx] = z;
    }
}

__global__ __launch_bounds__(256) void k_xm(const float* __restrict__ x,
                                            const int* __restrict__ mask,
                                            unsigned short* __restrict__ xm) {
    int idx = blockIdx.x * 256 + threadIdx.x;          // < 4*256*256*64
    int c = idx & 63, pix = idx >> 6;
    int iw = pix & 255, ih = (pix >> 8) & 255, b = pix >> 16;
    float v = (mask[pix] == 0) ? x[idx] : 0.0f;
    xm[((b * 258 + ih + 1) * 258 + iw + 1) * 64 + c] = f2bf_bits(v);
}

__global__ __launch_bounds__(256) void k_m1(const int* __restrict__ mask,
                                            unsigned char* __restrict__ m1) {
    int idx = blockIdx.x * 256 + threadIdx.x;          // < 4*128*128
    int ow = idx & 127, oh = (idx >> 7) & 127, b = idx >> 14;
    int act = 0;
    #pragma unroll
    for (int ky = 0; ky < 3; ++ky)
        #pragma unroll
        for (int kx = 0; kx < 3; ++kx) {
            int ih = 2 * oh - 1 + ky, iw = 2 * ow - 1 + kx;
            if ((unsigned)ih < 256u && (unsigned)iw < 256u)
                act |= (mask[(b * 256 + ih) * 256 + iw] == 0);
        }
    m1[idx] = (unsigned char)act;
}

// weight repack -> B-fragment layout: [t][nt][lane][e], k_in_chunk = (e&8)*2 + (lane/16)*8 + (e&7)
__global__ __launch_bounds__(256) void k_w1p(const float* __restrict__ w1,
                                             unsigned short* __restrict__ w1p) {
    int idx = blockIdx.x * 256 + threadIdx.x;          // < 18*8*32*16 = 73728
    int e = idx & 15, lane = (idx >> 4) & 31, nt = (idx >> 9) & 7, t = idx >> 12;
    int g = lane >> 4, n = nt * 16 + (lane & 15);
    int kk = ((e & 8) << 1) + g * 8 + (e & 7);
    int sp = t >> 1, c0 = (t & 1) << 5;                // sp = ky*3+kx
    int kidx = sp * 64 + c0 + kk;
    w1p[idx] = f2bf_bits(w1[kidx * 128 + n]);
}

__global__ __launch_bounds__(256) void k_w2p(const float* __restrict__ w2,
                                             unsigned short* __restrict__ w2p) {
    int idx = blockIdx.x * 256 + threadIdx.x;          // < 36*8*32*16 = 147456
    int e = idx & 15, lane = (idx >> 4) & 31, nt = (idx >> 9) & 7, t = idx >> 12;
    int g = lane >> 4, n = nt * 16 + (lane & 15);
    int kk = ((e & 8) << 1) + g * 8 + (e & 7);
    int sp = t >> 2, c0 = (t & 3) << 5;
    int kidx = sp * 128 + c0 + kk;
    w2p[idx] = f2bf_bits(w2[kidx * 128 + n]);
}

// ---------- WMMA implicit-GEMM convs (padded inputs, fully unrolled K) ----------
// wave tile: 16 output positions (along OW) x 64 out-channels (4 accumulators)
__global__ __launch_bounds__(128) void k_conv1(const unsigned short* __restrict__ xm,
                                               const unsigned short* __restrict__ w1p,
                                               float* __restrict__ raw) {
    int wid  = blockIdx.x * 4 + (threadIdx.x >> 5);    // 8192 waves
    int lane = threadIdx.x & 31;
    int ng  = wid & 1;
    int owt = (wid >> 1) & 7;
    int oh  = (wid >> 4) & 127;
    int b   = wid >> 11;
    int g = lane >> 4, l16 = lane & 15;
    int owBase = owt * 16;
    int ow = owBase + l16;
    // padded pixel (oh*2, ow*2) == unpadded (oh*2-1, ow*2-1)
    const unsigned short* abase =
        xm + (((size_t)(b * 258 + oh * 2) * 258 + ow * 2) * 64) + g * 8;
    const unsigned short* bbase = w1p + ng * 2048 + lane * 16;
    v8f acc[4];
    #pragma unroll
    for (int j = 0; j < 4; ++j) acc[j] = {};
    #pragma unroll
    for (int t = 0; t < 18; ++t) {
        const int sp = t >> 1, c0 = (t & 1) << 5;
        const int ky = sp / 3, kx = sp - ky * 3;
        const int aoff = (ky * 258 + kx) * 64 + c0;    // compile-time constant
        v16bf A = mk_a(abase + aoff);
        // stage all 4 B fragments first -> 8 loads in flight, WMMAs issue back-to-back
        us16 bb[4];
        #pragma unroll
        for (int j = 0; j < 4; ++j)
            bb[j] = *(const us16*)(bbase + t * 4096 + j * 512);
        #pragma unroll
        for (int j = 0; j < 4; ++j) {
            v16bf Bm = __builtin_bit_cast(v16bf, bb[j]);
            acc[j] = __builtin_amdgcn_wmma_f32_16x16x32_bf16(false, A, false, Bm,
                                                             (short)0, acc[j], false, false);
        }
    }
    #pragma unroll
    for (int j = 0; j < 4; ++j) {
        int n = (ng * 4 + j) * 16 + l16;
        #pragma unroll
        for (int r = 0; r < 8; ++r) {
            int m = r + g * 8;
            int pos = (b * 128 + oh) * 128 + owBase + m;
            raw[pos * 128 + n] = acc[j][r];
        }
    }
}

__global__ __launch_bounds__(128) void k_conv2(const unsigned short* __restrict__ y1,
                                               const unsigned short* __restrict__ w2p,
                                               float* __restrict__ raw) {
    int wid  = blockIdx.x * 4 + (threadIdx.x >> 5);    // 8192 waves
    int lane = threadIdx.x & 31;
    int ng  = wid & 1;
    int owt = (wid >> 1) & 7;
    int oh  = (wid >> 4) & 127;
    int b   = wid >> 11;
    int g = lane >> 4, l16 = lane & 15;
    int owBase = owt * 16;
    int ow = owBase + l16;
    // padded pixel (oh, ow) == unpadded (oh-1, ow-1)
    const unsigned short* abase =
        y1 + (((size_t)(b * 130 + oh) * 130 + ow) * 128) + g * 8;
    const unsigned short* bbase = w2p + ng * 2048 + lane * 16;
    v8f acc[4];
    #pragma unroll
    for (int j = 0; j < 4; ++j) acc[j] = {};
    #pragma unroll
    for (int t = 0; t < 36; ++t) {
        const int sp = t >> 2, c0 = (t & 3) << 5;
        const int ky = sp / 3, kx = sp - ky * 3;
        const int aoff = (ky * 130 + kx) * 128 + c0;   // compile-time constant
        v16bf A = mk_a(abase + aoff);
        us16 bb[4];
        #pragma unroll
        for (int j = 0; j < 4; ++j)
            bb[j] = *(const us16*)(bbase + t * 4096 + j * 512);
        #pragma unroll
        for (int j = 0; j < 4; ++j) {
            v16bf Bm = __builtin_bit_cast(v16bf, bb[j]);
            acc[j] = __builtin_amdgcn_wmma_f32_16x16x32_bf16(false, A, false, Bm,
                                                             (short)0, acc[j], false, false);
        }
    }
    #pragma unroll
    for (int j = 0; j < 4; ++j) {
        int n = (ng * 4 + j) * 16 + l16;
        #pragma unroll
        for (int r = 0; r < 8; ++r) {
            int m = r + g * 8;
            int pos = (b * 128 + oh) * 128 + owBase + m;
            raw[pos * 128 + n] = acc[j][r];
        }
    }
}

// ---------- masked BN: deterministic two-stage reduction ----------
__global__ __launch_bounds__(256) void k_stats(const float* __restrict__ raw,
                                               const unsigned char* __restrict__ m1,
                                               float* __restrict__ psum,
                                               float* __restrict__ psq,
                                               float* __restrict__ pcnt) {
    int c = threadIdx.x & 127;
    int half = threadIdx.x >> 7;
    int posBase = blockIdx.x * 64;                     // 1024 blocks x 64 positions
    float s = 0.f, q = 0.f, cnt = 0.f;
    for (int i = 0; i < 32; ++i) {
        int pos = posBase + half + i * 2;
        float mf = (float)m1[pos];
        float v  = raw[pos * 128 + c] * mf;
        s += v; q += v * v;
        if (c == 0) cnt += mf;
    }
    __shared__ float ss[256], sq[256], sc[2];
    ss[threadIdx.x] = s; sq[threadIdx.x] = q;
    if (c == 0) sc[half] = cnt;
    __syncthreads();
    if (threadIdx.x < 128) {
        psum[blockIdx.x * 128 + threadIdx.x] = ss[threadIdx.x] + ss[threadIdx.x + 128];
        psq [blockIdx.x * 128 + threadIdx.x] = sq[threadIdx.x] + sq[threadIdx.x + 128];
        if (threadIdx.x == 0) pcnt[blockIdx.x] = sc[0] + sc[1];
    }
}

__global__ __launch_bounds__(128) void k_finalize(const float* __restrict__ psum,
                                                  const float* __restrict__ psq,
                                                  const float* __restrict__ pcnt,
                                                  const float* __restrict__ gamma,
                                                  const float* __restrict__ beta,
                                                  float* __restrict__ ssb) {
    int c = threadIdx.x;                               // 128 threads, fixed-order sums
    float s = 0.f, q = 0.f, n = 0.f;
    for (int i = 0; i < 1024; ++i) {
        s += psum[i * 128 + c];
        q += psq [i * 128 + c];
        n += pcnt[i];
    }
    float mean = s / n;
    float var  = q / n - mean * mean;
    float rs   = rsqrtf(var + EPS);
    float sc   = gamma[c] * rs;
    ssb[c]       = sc;
    ssb[128 + c] = beta[c] - mean * sc;
}

__global__ __launch_bounds__(256) void k_apply1(const float* __restrict__ raw,
                                                const unsigned char* __restrict__ m1,
                                                const float* __restrict__ ssb,
                                                unsigned short* __restrict__ y1) {
    int idx = blockIdx.x * 256 + threadIdx.x;          // < 8,388,608
    int c = idx & 127, pos = idx >> 7;
    int ow = pos & 127, oh = (pos >> 7) & 127, b = pos >> 14;
    float r = raw[idx] * ssb[c] + ssb[128 + c];
    r = fmaxf(r, 0.0f);
    r = m1[pos] ? r : 0.0f;
    y1[((b * 130 + oh + 1) * 130 + ow + 1) * 128 + c] = f2bf_bits(r);
}

__global__ __launch_bounds__(256) void k_apply2(const float* __restrict__ raw,
                                                const unsigned char* __restrict__ m1,
                                                const float* __restrict__ ssb,
                                                float* __restrict__ out) {
    int idx = blockIdx.x * 256 + threadIdx.x;
    int c = idx & 127, pos = idx >> 7;
    float r = raw[idx] * ssb[c] + ssb[128 + c];
    out[idx] = m1[pos] ? r : 0.0f;
}

extern "C" void kernel_launch(void* const* d_in, const int* in_sizes, int n_in,
                              void* d_out, int out_size, void* d_ws, size_t ws_size,
                              hipStream_t stream) {
    const float* x      = (const float*)d_in[0];
    const int*   mask   = (const int*)  d_in[1];
    const float* w1     = (const float*)d_in[2];
    const float* gamma1 = (const float*)d_in[3];
    const float* beta1  = (const float*)d_in[4];
    const float* w2     = (const float*)d_in[5];
    const float* gamma2 = (const float*)d_in[6];
    const float* beta2  = (const float*)d_in[7];
    float* out = (float*)d_out;

    char* ws = (char*)d_ws;
    unsigned short* xm  = (unsigned short*)(ws + OFF_XM);
    float*          raw = (float*)         (ws + OFF_RAW);
    unsigned short* y1  = (unsigned short*)(ws + OFF_Y1);
    unsigned char*  m1  = (unsigned char*) (ws + OFF_M1);
    unsigned short* w1p = (unsigned short*)(ws + OFF_W1P);
    unsigned short* w2p = (unsigned short*)(ws + OFF_W2P);
    float*         psum = (float*)         (ws + OFF_PSUM);
    float*         psq  = (float*)         (ws + OFF_PSQ);
    float*         pcnt = (float*)         (ws + OFF_PCNT);
    float*         ssb  = (float*)         (ws + OFF_SS);

    // zero padded staging buffers (borders must be 0)
    k_zero16<<<8321, 256, 0, stream>>>((us8*)xm, 2130048);   // 34,080,768 B
    k_zero16<<<4225, 256, 0, stream>>>((us8*)y1, 1081600);   // 17,305,600 B

    // prep
    k_xm <<<65536, 256, 0, stream>>>(x, mask, xm);
    k_m1 <<<256,   256, 0, stream>>>(mask, m1);
    k_w1p<<<288,   256, 0, stream>>>(w1, w1p);
    k_w2p<<<576,   256, 0, stream>>>(w2, w2p);

    // conv1 -> BN1 -> ReLU -> y1 (bf16, padded)
    k_conv1   <<<2048, 128, 0, stream>>>(xm, w1p, raw);
    k_stats   <<<1024, 256, 0, stream>>>(raw, m1, psum, psq, pcnt);
    k_finalize<<<1,    128, 0, stream>>>(psum, psq, pcnt, gamma1, beta1, ssb);
    k_apply1  <<<32768,256, 0, stream>>>(raw, m1, ssb, y1);

    // conv2 -> BN2 -> out (fp32)
    k_conv2   <<<2048, 128, 0, stream>>>(y1, w2p, raw);
    k_stats   <<<1024, 256, 0, stream>>>(raw, m1, psum, psq, pcnt);
    k_finalize<<<1,    128, 0, stream>>>(psum, psq, pcnt, gamma2, beta2, ssb);
    k_apply2  <<<32768,256, 0, stream>>>(raw, m1, ssb, out);
}
